// DeepIceModelNoExtractor_86431921864760
// MI455X (gfx1250) — matrix operand order repaired and checked by reading, not compiled, not run on hardware
//
#include <hip/hip_runtime.h>
#include <math.h>

typedef _Float16 f16;
typedef _Float16 v16h __attribute__((ext_vector_type(16)));
typedef _Float16 v8h  __attribute__((ext_vector_type(8)));
typedef float    v8f  __attribute__((ext_vector_type(8)));

#define DIMC   384
#define HEADS  12
#define HD     32
#define MLPH   1536
#define BATCH  8
#define NSEQ   256
#define RS     288                 // padded sequence length (multiple of 32)
#define RTOT   (BATCH * RS)        // 2304 rows, 18 row-blocks of 128
#define NKT    18                  // 288/16 key tiles
#define NCH    9                   // 288/32 K-chunks for P*V
#define VP     304                 // V^T LDS row stride (f16) -> 32B aligned rows
#define PP     296                 // P LDS row stride (f16)   -> 16B aligned rows
#define NEGV   (-1.0e9f)
#define TSCALE 18.0f               // 30000/500*0.3
#define QSCALE 0.17677669529663687f
#define GK     64                  // GEMM K-step staged through LDS
#define SMEM_ATTN ((RS*HD + HD*VP + 8*16*PP) * 2)

__device__ __forceinline__ v16h cat8(v8h lo, v8h hi) {
  return __builtin_shufflevector(lo, hi, 0,1,2,3,4,5,6,7,8,9,10,11,12,13,14,15);
}
__device__ __forceinline__ v8f wmma16(v16h a, v16h b, v8f c) {
  return __builtin_amdgcn_wmma_f32_16x16x32_f16(false, a, false, b, (short)0, c, false, false);
}
__device__ __forceinline__ float gelu_exact(float x) {
  return 0.5f * x * (1.0f + erff(x * 0.70710678118654752f));
}
// effective mask value (0 or -1e9); padded positions -> -1e9; cls row -> 0
__device__ __forceinline__ float maskval(const float* mask, int b, int idx, int nvalid, int has_cls) {
  if (idx >= nvalid) return NEGV;
  if (has_cls) { if (idx == 0) return 0.0f; idx -= 1; }
  return mask[b * NSEQ + idx];
}
// async copy of 16B/lane from global (uniform base + per-lane offset) into LDS
__device__ __forceinline__ void async_copy_b128(unsigned lds_off, const void* gbase, unsigned voff) {
  asm volatile("global_load_async_to_lds_b128 %0, %1, %2"
               :: "v"(lds_off), "v"(voff), "s"(gbase) : "memory");
}
__device__ __forceinline__ void wait_async0() {
  asm volatile("s_wait_asynccnt 0x0" ::: "memory");
}

// ---------------------------------------------------------------- convert f32 -> f16
__global__ void cvt_kernel(const float* __restrict__ src, f16* __restrict__ dst, int n) {
  int i = blockIdx.x * 256 + threadIdx.x;
  if (i < n) dst[i] = (f16)src[i];
}

// ---------------------------------------------------------------- init padded x buffer
__global__ void init_x_kernel(const float* __restrict__ x, float* __restrict__ xb) {
  int row = blockIdx.x; int b = row / RS; int i = row % RS;
  float* o = xb + (size_t)row * DIMC;
  if (i < NSEQ) {
    const float* s = x + ((size_t)b * NSEQ + i) * DIMC;
    for (int c = threadIdx.x; c < DIMC; c += 128) o[c] = s[c];
  } else {
    for (int c = threadIdx.x; c < DIMC; c += 128) o[c] = 0.0f;
  }
}

// ---------------------------------------------------------------- prepend CLS token
__global__ void cls_kernel(const float* __restrict__ xin, const float* __restrict__ cls_w,
                           float* __restrict__ xo) {
  int row = blockIdx.x; int b = row / RS; int i = row % RS;
  float* o = xo + (size_t)row * DIMC;
  if (i == 0) {
    for (int c = threadIdx.x; c < DIMC; c += 128) o[c] = cls_w[c];
  } else if (i <= NSEQ) {
    const float* s = xin + ((size_t)b * RS + (i - 1)) * DIMC;
    for (int c = threadIdx.x; c < DIMC; c += 128) o[c] = s[c];
  } else {
    for (int c = threadIdx.x; c < DIMC; c += 128) o[c] = 0.0f;
  }
}

// ---------------------------------------------------------------- LayerNorm -> f16 (zeros padded rows)
__global__ void ln_kernel(const float* __restrict__ x, const float* __restrict__ g,
                          const float* __restrict__ bb, f16* __restrict__ xn, int nvalid) {
  int row = blockIdx.x; int i = row % RS; int tid = threadIdx.x;
  f16* o = xn + (size_t)row * DIMC;
  if (i >= nvalid) {
    for (int c = tid; c < DIMC; c += 128) o[c] = (f16)0.0f;
    return;
  }
  const float* xr = x + (size_t)row * DIMC;
  float s = 0.0f, s2 = 0.0f;
  for (int c = tid; c < DIMC; c += 128) { float v = xr[c]; s += v; s2 += v * v; }
  __shared__ float r1[128], r2[128];
  r1[tid] = s; r2[tid] = s2; __syncthreads();
  for (int st = 64; st > 0; st >>= 1) {
    if (tid < st) { r1[tid] += r1[tid + st]; r2[tid] += r2[tid + st]; }
    __syncthreads();
  }
  float mean = r1[0] * (1.0f / DIMC);
  float var  = r2[0] * (1.0f / DIMC) - mean * mean;
  float rstd = rsqrtf(var + 1e-5f);
  for (int c = tid; c < DIMC; c += 128)
    o[c] = (f16)((xr[c] - mean) * rstd * g[c] + bb[c]);
}

// ---------------------------------------------------------------- relative position bias (f16 [B,N,N,HD])
__global__ void rel_kernel(const float* __restrict__ x0, const float* __restrict__ w1d,
                           const float* __restrict__ b1d, const float* __restrict__ wp,
                           const float* __restrict__ bp, f16* __restrict__ rel) {
  int t = blockIdx.x * 256 + threadIdx.x;
  int b = t / (NSEQ * NSEQ); int rm = t % (NSEQ * NSEQ);
  int i = rm / NSEQ; int j = rm % NSEQ;
  const float* xi = x0 + ((size_t)b * NSEQ + i) * 4;
  const float* xj = x0 + ((size_t)b * NSEQ + j) * 4;
  float dp0 = xi[0] - xj[0], dp1 = xi[1] - xj[1], dp2 = xi[2] - xj[2];
  float dt  = xi[3] - xj[3] * TSCALE;
  float q0 = dp0 * dp0, q1 = dp1 * dp1, q2 = dp2 * dp2, qt = dt * dt;
  float ds2 = q0 + q1 + q2 - qt;
  float d = (ds2 >= 0.0f) ? sqrtf(ds2) : -sqrtf(-ds2);
  d += w1d[0]*dp0 + w1d[1]*dp1 + w1d[2]*dp2 + w1d[3]*dt
     + w1d[4]*q0  + w1d[5]*q1  + w1d[6]*q2  + w1d[7]*qt + b1d[0];
  float de = 1024.0f * fminf(4.0f, fmaxf(-4.0f, d));
  float emb[32];
  const float step = -0.57564627324851142f;  // -ln(10000)/16
  #pragma unroll
  for (int h = 0; h < 16; ++h) {
    float e = de * expf((float)h * step);
    emb[h] = sinf(e); emb[16 + h] = cosf(e);
  }
  f16* out = rel + (((size_t)(b * NSEQ + i) * NSEQ + j) << 5);
  #pragma unroll 4
  for (int c = 0; c < HD; ++c) {
    float s = bp[c];
    const float* wr = wp + c * HD;
    #pragma unroll
    for (int k = 0; k < 32; ++k) s += emb[k] * wr[k];
    out[c] = (f16)s;
  }
}

// ---------------------------------------------------------------- generic GEMM
// out = gamma * act(A(f16)[2304xK] * W(f16)[MxK]^T * scale + bias)
// W tile (64 x GK) staged through LDS with async global->LDS copies, double buffered.
__global__ __launch_bounds__(256)
void gemm_kernel(const f16* __restrict__ A, const f16* __restrict__ W,
                 const float* __restrict__ bias, const float* __restrict__ gamma,
                 float* __restrict__ outF, f16* __restrict__ outH,
                 int K, int M, float out_scale, int accum, int act) {
  __shared__ f16 Wt[2][64 * GK];     // 8 KB per buffer
  int tid = threadIdx.x;
  int wave = tid >> 5, lane = tid & 31;
  int m = lane & 15, half = lane >> 4;
  int row0 = blockIdx.y * 128 + wave * 16;
  int col0 = blockIdx.x * 64;
  const int nk = K / GK;

  // each thread copies 2 x 16B chunks of the 64xGK tile (512 chunks total)
  auto issue_tile = [&](int bi, int k0) {
    const f16* gb = W + (size_t)col0 * K + k0;
    #pragma unroll
    for (int u = 0; u < 2; ++u) {
      int idx = tid + u * 256;
      int rowi = idx >> 3, ch = idx & 7;
      unsigned voff = (unsigned)((rowi * K + ch * 8) * 2);
      unsigned loff = (unsigned)(uintptr_t)(&Wt[bi][rowi * GK + ch * 8]);
      async_copy_b128(loff, gb, voff);
    }
  };

  v8f acc[4] = {};
  issue_tile(0, 0);
  for (int kt = 0; kt < nk; ++kt) {
    wait_async0();
    __syncthreads();                       // tile kt resident for all waves
    if (kt + 1 < nk) issue_tile((kt + 1) & 1, (kt + 1) * GK);
    const f16* Wl = Wt[kt & 1];
    int k0 = kt * GK;
    #pragma unroll
    for (int ks = 0; ks < GK; ks += 32) {
      const f16* pa = A + (size_t)(row0 + m) * K + k0 + ks + half * 16;
      v16h a = cat8(*(const v8h*)pa, *(const v8h*)(pa + 8));
      #pragma unroll
      for (int f = 0; f < 4; ++f) {
        const f16* pb = Wl + (f * 16 + m) * GK + ks + half * 16;
        acc[f] = wmma16(a, *(const v16h*)pb, acc[f]);
      }
    }
    __syncthreads();                       // all waves done reading before overwrite
  }

  #pragma unroll
  for (int f = 0; f < 4; ++f) {
    int col = col0 + f * 16 + m;
    float bv = bias  ? bias[col]  : 0.0f;
    float gv = gamma ? gamma[col] : 1.0f;
    #pragma unroll
    for (int r = 0; r < 8; ++r) {
      int row = row0 + half * 8 + r;
      float v = acc[f][r] * out_scale + bv;
      if (act) v = gelu_exact(v);
      v *= gv;
      size_t idx = (size_t)row * M + col;
      if (outF) { if (accum) outF[idx] += v; else outF[idx] = v; }
      if (outH) outH[idx] = (f16)v;
    }
  }
}

// ---------------------------------------------------------------- fused attention per (batch, head)
__global__ __launch_bounds__(256)
void attn_kernel(const f16* __restrict__ qb, const f16* __restrict__ kbuf,
                 const f16* __restrict__ vbuf, f16* __restrict__ ob,
                 const float* __restrict__ mask, const f16* __restrict__ rel,
                 int nvalid, int has_cls) {
  extern __shared__ __align__(64) char smem[];
  f16* Kt   = (f16*)smem;              // [RS][HD]
  f16* Vt   = Kt + RS * HD;            // [HD][VP]  (transposed V)
  f16* Pall = Vt + HD * VP;            // [8][16][PP]
  int b = blockIdx.x / HEADS, h = blockIdx.x % HEADS;
  int tid = threadIdx.x, wave = tid >> 5, lane = tid & 31;
  int m = lane & 15, half = lane >> 4;

  for (int idx = tid; idx < RS * HD; idx += 256) {
    int j = idx >> 5, c = idx & 31;
    size_t g = (size_t)(b * RS + j) * DIMC + h * HD + c;
    Kt[idx]        = kbuf[g];
    Vt[c * VP + j] = vbuf[g];
  }
  __syncthreads();

  f16* Pw = Pall + wave * 16 * PP;
  const v8f vz = {};
  for (int t = wave; t < NKT; t += 8) {
    int q0 = t * 16;
    const f16* pq = qb + (size_t)(b * RS + q0 + m) * DIMC + h * HD + half * 16;
    v16h af = cat8(*(const v8h*)pq, *(const v8h*)(pq + 8));

    v8f sreg[NKT];
    #pragma unroll
    for (int jt = 0; jt < NKT; ++jt) {
      v16h bf = *(const v16h*)(Kt + (jt * 16 + m) * HD + half * 16);
      sreg[jt] = wmma16(af, bf, vz);
    }
    // bias + (layer-0) relative term
    #pragma unroll
    for (int jt = 0; jt < NKT; ++jt) {
      int j = jt * 16 + m;
      float mj = maskval(mask, b, j, nvalid, has_cls);
      #pragma unroll
      for (int r = 0; r < 8; ++r) {
        int i = q0 + half * 8 + r;
        float s = sreg[jt][r];
        if (j >= nvalid) {
          s = NEGV;
        } else {
          float mi = maskval(mask, b, i, nvalid, has_cls);
          float mx = fmaxf(mi, mj), mn = fminf(mi, mj);
          s += (mx < 0.0f) ? 0.0f : mn;
          if (rel && i < NSEQ && j < NSEQ) {
            const f16* rp = rel + (((size_t)(b * NSEQ + i) * NSEQ + j) << 5);
            const f16* qp = qb + (size_t)(b * RS + i) * DIMC + h * HD;
            float acc = 0.0f;
            #pragma unroll 8
            for (int c2 = 0; c2 < HD; ++c2) acc += (float)qp[c2] * (float)rp[c2];
            s += acc;
          }
        }
        sreg[jt][r] = s;
      }
    }
    // softmax across j (row spread over 16 lanes of each half + 18 tiles)
    float rinv[8];
    #pragma unroll
    for (int r = 0; r < 8; ++r) {
      float mx = -3.4e38f;
      #pragma unroll
      for (int jt = 0; jt < NKT; ++jt) mx = fmaxf(mx, sreg[jt][r]);
      for (int d2 = 1; d2 < 16; d2 <<= 1) mx = fmaxf(mx, __shfl_xor(mx, d2, 32));
      float sm = 0.0f;
      #pragma unroll
      for (int jt = 0; jt < NKT; ++jt) {
        float p = expf(sreg[jt][r] - mx);
        sreg[jt][r] = p; sm += p;
      }
      for (int d2 = 1; d2 < 16; d2 <<= 1) sm += __shfl_xor(sm, d2, 32);
      rinv[r] = 1.0f / sm;
    }
    // write P (f16) to per-wave LDS
    #pragma unroll
    for (int jt = 0; jt < NKT; ++jt)
      #pragma unroll
      for (int r = 0; r < 8; ++r)
        Pw[(half * 8 + r) * PP + jt * 16 + m] = (f16)(sreg[jt][r] * rinv[r]);
    asm volatile("s_wait_dscnt 0" ::: "memory");

    // O = P * V  (K chunks of 32)
    v8f oc0 = vz, oc1 = vz;
    #pragma unroll
    for (int ch = 0; ch < NCH; ++ch) {
      const f16* pp = Pw + m * PP + ch * 32 + half * 16;
      v16h pa  = cat8(*(const v8h*)pp, *(const v8h*)(pp + 8));
      v16h bv0 = *(const v16h*)(Vt + m * VP + ch * 32 + half * 16);
      v16h bv1 = *(const v16h*)(Vt + (16 + m) * VP + ch * 32 + half * 16);
      oc0 = wmma16(pa, bv0, oc0);
      oc1 = wmma16(pa, bv1, oc1);
    }
    // layer-0: O += P * rel
    if (rel) {
      #pragma unroll
      for (int r = 0; r < 8; ++r) {
        int i = q0 + half * 8 + r;
        if (i < NSEQ) {
          const f16* pr   = rel + (((size_t)(b * NSEQ + i) * NSEQ) << 5) + m;
          const f16* prow = Pw + (half * 8 + r) * PP;
          float a0 = 0.0f, a1 = 0.0f;
          for (int j = 0; j < NSEQ; ++j) {
            float pv = (float)prow[j];
            a0 += pv * (float)pr[(size_t)j * 32];
            a1 += pv * (float)pr[(size_t)j * 32 + 16];
          }
          oc0[r] += a0; oc1[r] += a1;
        }
      }
    }
    // store (zero padded rows)
    #pragma unroll
    for (int r = 0; r < 8; ++r) {
      int i = q0 + half * 8 + r;
      size_t go = (size_t)(b * RS + i) * DIMC + h * HD;
      if (i < nvalid) {
        ob[go + m]      = (f16)oc0[r];
        ob[go + 16 + m] = (f16)oc1[r];
      } else {
        ob[go + m]      = (f16)0.0f;
        ob[go + 16 + m] = (f16)0.0f;
      }
    }
  }
}

// ---------------------------------------------------------------- final head
__global__ void out_kernel(const float* __restrict__ xb, const float* __restrict__ out_w,
                           const float* __restrict__ out_b, float* __restrict__ out) {
  int t = threadIdx.x;
  if (t >= BATCH * 3) return;
  int b = t / 3, k = t % 3;
  const float* xr = xb + (size_t)b * RS * DIMC;
  float s = out_b[k];
  for (int d = 0; d < DIMC; ++d) s += xr[d] * out_w[k * DIMC + d];
  out[t] = s;
}

// ================================================================ host
extern "C" void kernel_launch(void* const* d_in, const int* in_sizes, int n_in,
                              void* d_out, int out_size, void* d_ws, size_t ws_size,
                              hipStream_t stream) {
  (void)in_sizes; (void)n_in; (void)out_size; (void)ws_size;
  const float* x      = (const float*)d_in[0];
  const float* x0     = (const float*)d_in[1];
  const float* mask   = (const float*)d_in[2];
  const float* rp_w1d = (const float*)d_in[3];
  const float* rp_b1d = (const float*)d_in[4];
  const float* rp_wp  = (const float*)d_in[5];
  const float* rp_bp  = (const float*)d_in[6];
  const float* sw[13]; for (int i = 0; i < 13; ++i) sw[i] = (const float*)d_in[7 + i];
  const float* bl[13]; for (int i = 0; i < 13; ++i) bl[i] = (const float*)d_in[20 + i];
  const float* bl_g1 = (const float*)d_in[33];
  const float* bl_g2 = (const float*)d_in[34];
  const float* cls_w = (const float*)d_in[35];
  const float* out_w = (const float*)d_in[36];
  const float* out_b = (const float*)d_in[37];

  // workspace carve (all slot sizes are multiples of 256B)
  char* p = (char*)d_ws;
  f16*  rel16 = (f16*)p;  p += (size_t)BATCH * NSEQ * NSEQ * HD * 2;   // 33.5 MB
  float* xb   = (float*)p; p += (size_t)RTOT * DIMC * 4;
  float* xb2  = (float*)p; p += (size_t)RTOT * DIMC * 4;
  f16*  xn16 = (f16*)p;   p += (size_t)RTOT * DIMC * 2;
  f16*  qb   = (f16*)p;   p += (size_t)RTOT * DIMC * 2;
  f16*  kb   = (f16*)p;   p += (size_t)RTOT * DIMC * 2;
  f16*  vb   = (f16*)p;   p += (size_t)RTOT * DIMC * 2;
  f16*  obuf = (f16*)p;   p += (size_t)RTOT * DIMC * 2;
  f16*  hb   = (f16*)p;   p += (size_t)RTOT * MLPH * 2;
  f16*  wq16 = (f16*)p;   p += (size_t)DIMC * DIMC * 2;
  f16*  wk16 = (f16*)p;   p += (size_t)DIMC * DIMC * 2;
  f16*  wv16 = (f16*)p;   p += (size_t)DIMC * DIMC * 2;
  f16*  wo16 = (f16*)p;   p += (size_t)DIMC * DIMC * 2;
  f16*  w116 = (f16*)p;   p += (size_t)MLPH * DIMC * 2;
  f16*  w216 = (f16*)p;   p += (size_t)DIMC * MLPH * 2;

  auto cvt = [&](const float* s, f16* d, int n) {
    cvt_kernel<<<(n + 255) / 256, 256, 0, stream>>>(s, d, n);
  };
  auto gemm = [&](const f16* A, const f16* W, const float* bias, const float* gamma,
                  float* oF, f16* oH, int K, int M, float sc, int accum, int act) {
    dim3 g(M / 64, RTOT / 128);
    gemm_kernel<<<g, 256, 0, stream>>>(A, W, bias, gamma, oF, oH, K, M, sc, accum, act);
  };
  auto layer = [&](const float** w, int l, const float* g1, const float* g2,
                   const f16* relp, int nvalid, int has_cls, float* xc) {
    const float* ln1g = w[0] + (size_t)l * DIMC;
    const float* ln1b = w[1] + (size_t)l * DIMC;
    const float* ln2g = w[2] + (size_t)l * DIMC;
    const float* ln2b = w[3] + (size_t)l * DIMC;
    cvt(w[4] + (size_t)l * DIMC * DIMC, wq16, DIMC * DIMC);
    cvt(w[5] + (size_t)l * DIMC * DIMC, wk16, DIMC * DIMC);
    cvt(w[6] + (size_t)l * DIMC * DIMC, wv16, DIMC * DIMC);
    cvt(w[7] + (size_t)l * DIMC * DIMC, wo16, DIMC * DIMC);
    cvt(w[9] + (size_t)l * MLPH * DIMC, w116, MLPH * DIMC);
    cvt(w[11] + (size_t)l * DIMC * MLPH, w216, DIMC * MLPH);
    const float* bo = w[8]  + (size_t)l * DIMC;
    const float* b1 = w[10] + (size_t)l * MLPH;
    const float* b2 = w[12] + (size_t)l * DIMC;

    ln_kernel<<<RTOT, 128, 0, stream>>>(xc, ln1g, ln1b, xn16, nvalid);
    gemm(xn16, wq16, nullptr, nullptr, nullptr, qb, DIMC, DIMC, QSCALE, 0, 0);
    gemm(xn16, wk16, nullptr, nullptr, nullptr, kb, DIMC, DIMC, 1.0f, 0, 0);
    gemm(xn16, wv16, nullptr, nullptr, nullptr, vb, DIMC, DIMC, 1.0f, 0, 0);
    attn_kernel<<<BATCH * HEADS, 256, SMEM_ATTN, stream>>>(qb, kb, vb, obuf, mask,
                                                           relp, nvalid, has_cls);
    gemm(obuf, wo16, bo, g1, xc, nullptr, DIMC, DIMC, 1.0f, 1, 0);
    ln_kernel<<<RTOT, 128, 0, stream>>>(xc, ln2g, ln2b, xn16, nvalid);
    gemm(xn16, w116, b1, nullptr, nullptr, hb, DIMC, MLPH, 1.0f, 0, 1);
    gemm(hb, w216, b2, g2, xc, nullptr, MLPH, DIMC, 1.0f, 1, 0);
  };

  rel_kernel<<<(BATCH * NSEQ * NSEQ) / 256, 256, 0, stream>>>(x0, rp_w1d, rp_b1d,
                                                              rp_wp, rp_bp, rel16);
  init_x_kernel<<<RTOT, 128, 0, stream>>>(x, xb);

  for (int l = 0; l < 4; ++l)
    layer(sw, l, nullptr, nullptr, (l == 0) ? rel16 : nullptr, NSEQ, 0, xb);

  cls_kernel<<<RTOT, 128, 0, stream>>>(xb, cls_w, xb2);

  for (int l = 0; l < 12; ++l)
    layer(bl, l, bl_g1 + (size_t)l * DIMC, bl_g2 + (size_t)l * DIMC,
          nullptr, NSEQ + 1, 1, xb2);

  out_kernel<<<1, 32, 0, stream>>>(xb2, out_w, out_b, (float*)d_out);
}
